// MultiheadAttention_25125558681777
// MI455X (gfx1250) — matrix operand order repaired and checked by reading, not compiled, not run on hardware
//
#include <hip/hip_runtime.h>
#include <hip/hip_bf16.h>

#define EMB   768
#define HEADS 12
#define HD    64
#define BB    8
#define TT    1024
#define MROWS (BB*TT)      // 8192
#define NQKV  (3*EMB)      // 2304

typedef __bf16 bf16;
typedef bf16  bf16x8  __attribute__((ext_vector_type(8)));
typedef bf16  bf16x16 __attribute__((ext_vector_type(16)));
typedef float f32x8   __attribute__((ext_vector_type(8)));

__device__ __forceinline__ bf16x16 make16(bf16x8 lo, bf16x8 hi) {
  bf16x16 r;
#pragma unroll
  for (int i = 0; i < 8; ++i) { r[i] = lo[i]; r[i + 8] = hi[i]; }
  return r;
}

// A-matrix fragment (16x32 bf16): lane<16 holds K 0..7 (v0..3) + 16..23 (v4..7),
// lane>=16 holds K 8..15 + 24..31.  `row` points at 32 contiguous K values.
__device__ __forceinline__ bf16x16 frag_a32(const bf16* row, int lane) {
  const int kb = (lane & 16) ? 8 : 0;
  bf16x8 lo = *(const bf16x8*)(row + kb);
  bf16x8 hi = *(const bf16x8*)(row + kb + 16);
  return make16(lo, hi);
}

// B-matrix fragment (32x16 bf16): lanes 0-15 hold K 0..7 + 16..23, lanes 16-31 K 8..15 + 24..31.
// `col` points at 32 contiguous K values of this lane's N column.
__device__ __forceinline__ bf16x16 frag_b32(const bf16* col, int lane) {
  const int kb = (lane & 16) ? 8 : 0;
  bf16x8 lo = *(const bf16x8*)(col + kb);
  bf16x8 hi = *(const bf16x8*)(col + kb + 16);
  return make16(lo, hi);
}

#define WMMA_BF16(A, B, C) \
  __builtin_amdgcn_wmma_f32_16x16x32_bf16(false, (A), false, (B), (short)0, (C), false, false)

// ---------------------------------------------------------------------------
// Generic 128x128-tile GEMM, bf16 WMMA, f32 accumulate.
// MODE 0: A = f32 x, epilogue scatters bf16 into q[bh][t][d] (pre-scaled by
//         1/sqrt(HD)), k[bh][t][d], vT[bh][d][t]
// MODE 1: A = bf16 attn_out, epilogue writes f32 out = A@W + bias
// ---------------------------------------------------------------------------
template <int MODE>
__global__ __launch_bounds__(256) void gemm_kernel(
    const float* __restrict__ Af32, const bf16* __restrict__ Abf,
    const float* __restrict__ Bw, const float* __restrict__ bias,
    bf16* __restrict__ qbuf, bf16* __restrict__ kbuf, bf16* __restrict__ vT,
    float* __restrict__ outf, int Ncols) {
  __shared__ bf16 As[128 * 32];   // [row][k]
  __shared__ bf16 Bt[128 * 32];   // [n][k]  (transposed during staging)

  const int tid  = threadIdx.x;
  const int lane = tid & 31;
  const int wave = tid >> 5;
  const int m0 = blockIdx.y * 128;
  const int n0 = blockIdx.x * 128;
  const int wm = (wave & 3) * 32;   // wave covers rows wm..wm+31
  const int wn = (wave >> 2) * 64;  // wave covers cols wn..wn+63

  f32x8 acc[2][4];
#pragma unroll
  for (int i = 0; i < 2; ++i)
#pragma unroll
    for (int j = 0; j < 4; ++j) {
      f32x8 z = {0.f, 0.f, 0.f, 0.f, 0.f, 0.f, 0.f, 0.f};
      acc[i][j] = z;
    }

  for (int kk = 0; kk < EMB; kk += 32) {
    // ---- stage A tile (128x32) into LDS as bf16 ----
    {
      const int r = tid >> 1;
      const int c = (tid & 1) * 16;
      if (MODE == 0) {
        const float* src = Af32 + (size_t)(m0 + r) * EMB + kk + c;
        bf16x8 v0, v1;
#pragma unroll
        for (int j = 0; j < 8; ++j) { v0[j] = (bf16)src[j]; v1[j] = (bf16)src[8 + j]; }
        *(bf16x8*)(As + r * 32 + c)     = v0;
        *(bf16x8*)(As + r * 32 + c + 8) = v1;
      } else {
        const bf16* src = Abf + (size_t)(m0 + r) * EMB + kk + c;
        *(bf16x8*)(As + r * 32 + c)     = *(const bf16x8*)src;
        *(bf16x8*)(As + r * 32 + c + 8) = *(const bf16x8*)(src + 8);
      }
    }
    // ---- stage B tile (32x128) transposed into LDS [n][k] ----
    {
      const int k  = tid >> 3;
      const int nc = (tid & 7) * 16;
      const float* src = Bw + (size_t)(kk + k) * Ncols + n0 + nc;
#pragma unroll
      for (int j = 0; j < 16; ++j) Bt[(nc + j) * 32 + k] = (bf16)src[j];
    }
    __syncthreads();

    bf16x16 af[2], bf_[4];
#pragma unroll
    for (int i = 0; i < 2; ++i)
      af[i] = frag_a32(As + (wm + i * 16 + (lane & 15)) * 32, lane);
#pragma unroll
    for (int j = 0; j < 4; ++j)
      bf_[j] = frag_b32(Bt + (wn + j * 16 + (lane & 15)) * 32, lane);
    // Barrier directly after the fragment reads: next iteration's global loads
    // and LDS stores can overlap with this iteration's WMMAs.
    __syncthreads();
#pragma unroll
    for (int i = 0; i < 2; ++i)
#pragma unroll
      for (int j = 0; j < 4; ++j) acc[i][j] = WMMA_BF16(af[i], bf_[j], acc[i][j]);
  }

  // ---- epilogue ----
  const int mhi = (lane & 16) ? 8 : 0;
  const int nl  = lane & 15;
#pragma unroll
  for (int i = 0; i < 2; ++i) {
#pragma unroll
    for (int j = 0; j < 4; ++j) {
      const int N = n0 + wn + j * 16 + nl;
      const float bv = bias[N];
#pragma unroll
      for (int e = 0; e < 8; ++e) {
        const int M = m0 + wm + i * 16 + mhi + e;
        float val = acc[i][j][e] + bv;
        if (MODE == 0) {
          const int which = N / EMB;          // 0=q 1=k 2=v
          const int rem = N - which * EMB;
          const int h = rem >> 6;
          const int d = rem & 63;
          const int b = M >> 10;
          const int t = M & 1023;
          const int bh = b * HEADS + h;
          if (which == 0) {
            // fold the attention scale 1/sqrt(64) into Q (exact: exponent shift)
            qbuf[((size_t)bh * TT + t) * HD + d] = (bf16)(val * 0.125f);
          } else if (which == 1) {
            kbuf[((size_t)bh * TT + t) * HD + d] = (bf16)val;
          } else {
            vT[((size_t)bh * HD + d) * TT + t]   = (bf16)val;  // transposed
          }
        } else {
          outf[(size_t)M * EMB + N] = val;
        }
      }
    }
  }
}

// ---------------------------------------------------------------------------
// Flash-attention: one wave per 16-query tile; 32-key blocks; causal.
// S = Q K^T (2 tiles x 2 K-chunks = 4 WMMA), online softmax, O += P V (4 WMMA).
// Mask-free main loop; single diagonal block peeled with masking.
// ---------------------------------------------------------------------------
__global__ __launch_bounds__(256) void attn_kernel(const bf16* __restrict__ q,
                                                   const bf16* __restrict__ k,
                                                   const bf16* __restrict__ vT,
                                                   bf16* __restrict__ attnOut) {
  __shared__ bf16 pAll[8][16 * 32];   // per-wave P bounce buffer (C-layout -> A-layout)
  const int tid  = threadIdx.x;
  const int lane = tid & 31;
  const int wave = tid >> 5;
  bf16* pBuf = pAll[wave];

  const int tile = blockIdx.x * 8 + wave;   // 6144 tiles total
  const int bh = tile >> 6;                 // batch*head
  const int q0 = (tile & 63) << 4;          // query tile start

  const bf16* qBase = q  + (size_t)bh * TT * HD;
  const bf16* kBase = k  + (size_t)bh * TT * HD;
  const bf16* vBase = vT + (size_t)bh * HD * TT;

  const int rl  = lane & 15;
  const int mhi = (lane & 16) ? 8 : 0;

  const bf16x16 qa0 = frag_a32(qBase + (q0 + rl) * HD, lane);        // d 0..31
  const bf16x16 qa1 = frag_a32(qBase + (q0 + rl) * HD + 32, lane);   // d 32..63

  float m[8], l[8];
  f32x8 o[4];
#pragma unroll
  for (int e = 0; e < 8; ++e) { m[e] = -3.0e38f; l[e] = 0.f; }
#pragma unroll
  for (int t = 0; t < 4; ++t) {
    f32x8 z = {0.f, 0.f, 0.f, 0.f, 0.f, 0.f, 0.f, 0.f};
    o[t] = z;
  }

  // Exactly one partially-masked (diagonal) 32-key block per 16-query tile.
  const int lastBlk = (q0 + 15) >> 5;

  auto body = [&](int key0, bool diag) {
    // ---- scores (Q pre-scaled by 1/sqrt(HD)) ----
    f32x8 s[2];
#pragma unroll
    for (int nt = 0; nt < 2; ++nt) {
      const bf16* kr = kBase + (key0 + nt * 16 + rl) * HD;   // this lane's key row
      f32x8 c = {0.f, 0.f, 0.f, 0.f, 0.f, 0.f, 0.f, 0.f};
      c = WMMA_BF16(qa0, frag_b32(kr, lane), c);
      c = WMMA_BF16(qa1, frag_b32(kr + 32, lane), c);
      s[nt] = c;
    }
    if (diag) {   // causal mask, only for the diagonal block
#pragma unroll
      for (int nt = 0; nt < 2; ++nt) {
        const int kg = key0 + nt * 16 + rl;
#pragma unroll
        for (int e = 0; e < 8; ++e) {
          if (kg > q0 + mhi + e) s[nt][e] = -3.0e38f;
        }
      }
    }
    // ---- online softmax (row stats across the 16-lane N group) ----
    float p0[8], p1[8], alpha[8];
#pragma unroll
    for (int e = 0; e < 8; ++e) {
      float bm = fmaxf(s[0][e], s[1][e]);
#pragma unroll
      for (int off = 8; off; off >>= 1) bm = fmaxf(bm, __shfl_xor(bm, off, 16));
      const float mn = fmaxf(m[e], bm);
      alpha[e] = __expf(m[e] - mn);
      m[e] = mn;
      p0[e] = __expf(s[0][e] - mn);
      p1[e] = __expf(s[1][e] - mn);
      float rs = p0[e] + p1[e];
#pragma unroll
      for (int off = 8; off; off >>= 1) rs += __shfl_xor(rs, off, 16);
      l[e] = l[e] * alpha[e] + rs;
    }
#pragma unroll
    for (int t = 0; t < 4; ++t)
#pragma unroll
      for (int e = 0; e < 8; ++e) o[t][e] *= alpha[e];
    // ---- P: C-layout -> A-layout via per-wave LDS (DS ops in-order per wave) ----
#pragma unroll
    for (int e = 0; e < 8; ++e) {
      const int row = mhi + e;
      pBuf[row * 32 + rl]      = (bf16)p0[e];
      pBuf[row * 32 + rl + 16] = (bf16)p1[e];
    }
    const bf16x16 pa = frag_a32(pBuf + rl * 32, lane);
    // ---- O += P @ V  (V pre-transposed: contiguous keys per d column) ----
#pragma unroll
    for (int t = 0; t < 4; ++t) {
      const bf16* vc = vBase + (size_t)(t * 16 + rl) * TT + key0;
      o[t] = WMMA_BF16(pa, frag_b32(vc, lane), o[t]);
    }
  };

  for (int blk = 0; blk < lastBlk; ++blk) body(blk * 32, false);  // mask-free
  body(lastBlk * 32, true);                                       // diagonal

  // ---- finalize: multiply by 1/rowsum, store bf16 [B][T][C] for projection ----
  const int b = bh / HEADS, h = bh % HEADS;
  float inv[8];
#pragma unroll
  for (int e = 0; e < 8; ++e) inv[e] = 1.0f / l[e];
#pragma unroll
  for (int t = 0; t < 4; ++t)
#pragma unroll
    for (int e = 0; e < 8; ++e) {
      const int M = q0 + mhi + e;
      attnOut[(size_t)(b * TT + M) * EMB + h * HD + t * 16 + rl] =
          (bf16)(o[t][e] * inv[e]);
    }
}

extern "C" void kernel_launch(void* const* d_in, const int* in_sizes, int n_in,
                              void* d_out, int out_size, void* d_ws, size_t ws_size,
                              hipStream_t stream) {
  const float* x      = (const float*)d_in[0];
  const float* W_attn = (const float*)d_in[1];
  const float* b_attn = (const float*)d_in[2];
  const float* W_proj = (const float*)d_in[3];
  const float* b_proj = (const float*)d_in[4];
  float* out = (float*)d_out;

  const size_t headElems = (size_t)BB * HEADS * TT * HD;  // 6,291,456
  bf16* qbuf    = (bf16*)d_ws;
  bf16* kbuf    = qbuf + headElems;
  bf16* vT      = kbuf + headElems;
  bf16* attnOut = vT + headElems;   // [B*T][EMB] bf16

  dim3 g1(NQKV / 128, MROWS / 128);   // 18 x 64
  gemm_kernel<0><<<g1, 256, 0, stream>>>(x, nullptr, W_attn, b_attn,
                                         qbuf, kbuf, vT, nullptr, NQKV);

  attn_kernel<<<(BB * HEADS * (TT / 16)) / 8, 256, 0, stream>>>(qbuf, kbuf, vT, attnOut);

  dim3 g2(EMB / 128, MROWS / 128);    // 6 x 64
  gemm_kernel<1><<<g2, 256, 0, stream>>>(nullptr, attnOut, W_proj, b_proj,
                                         nullptr, nullptr, nullptr, out, EMB);
}
